// GMTConvAttention_52845277610103
// MI455X (gfx1250) — compile-verified, matrix operand order checked by reading, not run on hardware
//
#include <hip/hip_runtime.h>
#include <hip/hip_bf16.h>
#include <stdint.h>

#define NN  50000
#define EE  800000
#define DD  256
#define HH  8
#define DHD 32
#define SS  32
#define EPSV 1e-5f

#if defined(__gfx1250__)
#define USE_ASYNC_LDS 1
#endif

typedef __attribute__((ext_vector_type(16))) __bf16 v16bf;
typedef __attribute__((ext_vector_type(8)))  float  v8f;

union Frag16 { uint4 u[2]; v16bf v; };

__device__ __forceinline__ unsigned short f2bf(float x) {
  unsigned u = __float_as_uint(x);
  unsigned r = u + 0x7FFFu + ((u >> 16) & 1u);   // round-to-nearest-even
  return (unsigned short)(r >> 16);
}

// ---------------------------------------------------------------- precision prep
__global__ void __launch_bounds__(256)
conv_x_bf16(const float* __restrict__ x, unsigned short* __restrict__ xb) {
  const size_t i = (size_t)blockIdx.x * DD + threadIdx.x;
  xb[i] = f2bf(x[i]);
}

// W[h][d][f] (f32) -> WT[n=h*32+f][k=d] (bf16), so B-fragments are contiguous.
__global__ void __launch_bounds__(256)
prep_weights(const float* __restrict__ Wk, const float* __restrict__ Wv,
             unsigned short* __restrict__ wkT, unsigned short* __restrict__ wvT) {
  const int n = blockIdx.x, k = threadIdx.x;
  const int h = n >> 5, f = n & 31;
  const size_t src = (size_t)h * DD * DHD + (size_t)k * DHD + f;
  if (blockIdx.y == 0) wkT[(size_t)n * DD + k] = f2bf(Wk[src]);
  else                 wvT[(size_t)n * DD + k] = f2bf(Wv[src]);
}

// ---------------------------------------------------------------- degree / norm
__global__ void __launch_bounds__(256)
deg_init(float* __restrict__ deg) {
  const int i = blockIdx.x * 256 + threadIdx.x;
  if (i < NN) deg[i] = 1.0f;               // self-loop contribution
}

__global__ void __launch_bounds__(256)
deg_count(const int* __restrict__ ei, float* __restrict__ deg) {
  const int e = blockIdx.x * 256 + threadIdx.x;
  if (e < EE) atomicAdd(&deg[ei[EE + e]], 1.0f);   // dst row
}

__global__ void __launch_bounds__(256)
deg_inv(const float* __restrict__ deg, float* __restrict__ dinv) {
  const int i = blockIdx.x * 256 + threadIdx.x;
  if (i < NN) dinv[i] = rsqrtf(fmaxf(deg[i], 1.0f));
}

// ---------------------------------------------------------------- fused K/V GEMM (WMMA bf16)
// hk = x @ WkT^T, hv = x @ WvT^T ; per block: one 16-row M tile, 8 waves x 2 N-tiles.
__global__ void __launch_bounds__(256)
gemm_kv_wmma(const unsigned short* __restrict__ xb,
             const unsigned short* __restrict__ wkT,
             const unsigned short* __restrict__ wvT,
             float* __restrict__ hk, float* __restrict__ hv) {
  __shared__ __align__(16) unsigned short As[16 * DD];   // 8 KB A tile (16 x 256 bf16)
  const int t = threadIdx.x;
  const int mtile = blockIdx.x;

#ifdef USE_ASYNC_LDS
  { // stage A tile via async global->LDS DMA (ASYNCcnt-tracked, no VGPR round-trip)
    const uint4* src = (const uint4*)(xb + (size_t)mtile * 16 * DD);
    // low 32 bits of a generic shared-space address == LDS byte offset (flat aperture rule)
    const unsigned ldsbase = (unsigned)(uintptr_t)(const void*)As;
    for (int j = t; j < (16 * DD) / 8; j += 256) {
      unsigned lds_off = ldsbase + (unsigned)j * 16u;
      unsigned long long gaddr = (unsigned long long)(uintptr_t)(src + j);
      asm volatile("global_load_async_to_lds_b128 %0, %1, off"
                   :: "v"(lds_off), "v"(gaddr) : "memory");
    }
    asm volatile("s_wait_asynccnt 0x0" ::: "memory");
  }
#else
  { // fallback: stage A tile with regular contiguous 16B copies
    const uint4* src = (const uint4*)(xb + (size_t)mtile * 16 * DD);
    uint4* dst = (uint4*)As;
    for (int j = t; j < (16 * DD) / 8; j += 256) dst[j] = src[j];
  }
#endif
  __syncthreads();

  const int wave = t >> 5, lane = t & 31;
  const int m = lane & 15, koff = (lane >> 4) * 8;

  v8f ck[2] = {};
  v8f cv[2] = {};

  for (int ks = 0; ks < DD / 32; ++ks) {
    const int k0 = ks * 32;
    Frag16 a;
    a.u[0] = *(const uint4*)(As + m * DD + k0 + koff);
    a.u[1] = *(const uint4*)(As + m * DD + k0 + 16 + koff);
#pragma unroll
    for (int j = 0; j < 2; ++j) {
      const int ncol = (wave * 2 + j) * 16 + (lane & 15);
      Frag16 b;
      b.u[0] = *(const uint4*)(wkT + (size_t)ncol * DD + k0 + koff);
      b.u[1] = *(const uint4*)(wkT + (size_t)ncol * DD + k0 + 16 + koff);
      ck[j] = __builtin_amdgcn_wmma_f32_16x16x32_bf16(false, a.v, false, b.v,
                                                      (short)0, ck[j], false, false);
      b.u[0] = *(const uint4*)(wvT + (size_t)ncol * DD + k0 + koff);
      b.u[1] = *(const uint4*)(wvT + (size_t)ncol * DD + k0 + 16 + koff);
      cv[j] = __builtin_amdgcn_wmma_f32_16x16x32_bf16(false, a.v, false, b.v,
                                                      (short)0, cv[j], false, false);
    }
  }

#pragma unroll
  for (int j = 0; j < 2; ++j) {
    const int ncol = (wave * 2 + j) * 16 + (lane & 15);
#pragma unroll
    for (int r = 0; r < 8; ++r) {
      const int row = mtile * 16 + r + 8 * (lane >> 4);
      hk[(size_t)row * DD + ncol] = ck[j][r];
      hv[(size_t)row * DD + ncol] = cv[j][r];
    }
  }
}

// ---------------------------------------------------------------- GCN aggregate
// init with bias + self-loop message, then edge atomics.
__global__ void __launch_bounds__(256)
kv_init(const float* __restrict__ hk, const float* __restrict__ hv,
        const float* __restrict__ bk, const float* __restrict__ bv,
        const float* __restrict__ dinv,
        float* __restrict__ kacc, float* __restrict__ vacc) {
  const int n = blockIdx.x, t = threadIdx.x;
  const float w = dinv[n] * dinv[n];
  const size_t i = (size_t)n * DD + t;
  kacc[i] = bk[t] + hk[i] * w;
  vacc[i] = bv[t] + hv[i] * w;
}

__global__ void __launch_bounds__(256)
spmm_edges(const int* __restrict__ ei,
           const float* __restrict__ hk, const float* __restrict__ hv,
           const float* __restrict__ dinv,
           float* __restrict__ kacc, float* __restrict__ vacc) {
  const int e = blockIdx.x, t = threadIdx.x;
  const int s = ei[e], d = ei[EE + e];
  const float w = dinv[s] * dinv[d];
  atomicAdd(&kacc[(size_t)d * DD + t], hk[(size_t)s * DD + t] * w);
  atomicAdd(&vacc[(size_t)d * DD + t], hv[(size_t)s * DD + t] * w);
}

__global__ void __launch_bounds__(256)
conv_k_bf16(const float* __restrict__ kacc, unsigned short* __restrict__ kb) {
  const size_t i = (size_t)blockIdx.x * DD + threadIdx.x;
  kb[i] = f2bf(kacc[i]);
}

// ---------------------------------------------------------------- q projection (scaled, bf16)
__global__ void __launch_bounds__(256)
qproj(const float* __restrict__ seed, const float* __restrict__ q_w,
      unsigned short* __restrict__ qb) {
  const int s = blockIdx.x, t = threadIdx.x;
  float acc = 0.f;
  for (int k = 0; k < DD; ++k) acc += seed[(size_t)s * DD + k] * q_w[(size_t)t * DD + k];
  qb[(size_t)s * DD + t] = f2bf(acc * 0.17677669529663687f);   // DH^-0.5
}

// ---------------------------------------------------------------- attention scores (WMMA bf16)
// scores[h][s][n] = q_h @ k_h^T ; K=DH=32 -> one WMMA per 16x16 tile.
__global__ void __launch_bounds__(256)
scores_wmma(const unsigned short* __restrict__ qb,
            const unsigned short* __restrict__ kb,
            float* __restrict__ scores) {
  const int t = threadIdx.x, wave = t >> 5, lane = t & 31;
  const int ntile = blockIdx.x * 8 + wave;      // wave-uniform guard: EXEC stays full
  if (ntile >= NN / 16) return;
  const int h = blockIdx.y;
  const int koff = (lane >> 4) * 8;
  const int ncol = ntile * 16 + (lane & 15);

  Frag16 b;   // k-tile fragment, reused for both S tiles
  b.u[0] = *(const uint4*)(kb + (size_t)ncol * DD + h * DHD + koff);
  b.u[1] = *(const uint4*)(kb + (size_t)ncol * DD + h * DHD + 16 + koff);

#pragma unroll
  for (int st = 0; st < 2; ++st) {
    const int srow = st * 16 + (lane & 15);
    Frag16 a;
    a.u[0] = *(const uint4*)(qb + (size_t)srow * DD + h * DHD + koff);
    a.u[1] = *(const uint4*)(qb + (size_t)srow * DD + h * DHD + 16 + koff);
    v8f c = {};
    c = __builtin_amdgcn_wmma_f32_16x16x32_bf16(false, a.v, false, b.v,
                                                (short)0, c, false, false);
#pragma unroll
    for (int r = 0; r < 8; ++r) {
      const int s = st * 16 + r + 8 * (lane >> 4);
      scores[((size_t)(h * SS + s)) * NN + ncol] = c[r];
    }
  }
}

// ---------------------------------------------------------------- softmax + ctx (fused)
__global__ void __launch_bounds__(256)
softmax_ctx(const float* __restrict__ scores, const float* __restrict__ vacc,
            float* __restrict__ ctx) {
  __shared__ float red[256];
  const int t = threadIdx.x;
  const int h = blockIdx.x >> 5, s = blockIdx.x & 31;
  const float* row = scores + ((size_t)(h * SS + s)) * NN;

  float lm = -3.4e38f;
  for (int n = t; n < NN; n += 256) lm = fmaxf(lm, row[n]);
  red[t] = lm; __syncthreads();
  for (int off = 128; off; off >>= 1) { if (t < off) red[t] = fmaxf(red[t], red[t + off]); __syncthreads(); }
  const float mx = red[0]; __syncthreads();

  float ls = 0.f;
  for (int n = t; n < NN; n += 256) ls += __expf(row[n] - mx);
  red[t] = ls; __syncthreads();
  for (int off = 128; off; off >>= 1) { if (t < off) red[t] += red[t + off]; __syncthreads(); }
  const float denom = red[0]; __syncthreads();

  const int d = t & 31, chunk = t >> 5;
  float acc = 0.f;
  for (int n = chunk; n < NN; n += 8) {
    __builtin_prefetch(row + n + 4096, 0, 0);
    acc += __expf(row[n] - mx) * vacc[(size_t)n * DD + h * DHD + d];
  }
  red[t] = acc; __syncthreads();
  if (t < 32) {
    float sum = 0.f;
#pragma unroll
    for (int c2 = 0; c2 < 8; ++c2) sum += red[c2 * 32 + t];
    ctx[(size_t)s * DD + h * DHD + t] = sum / denom;
  }
}

// ---------------------------------------------------------------- epilogue: proj + 2x LN + FF
__global__ void __launch_bounds__(256)
epilogue(const float* __restrict__ ctx, const float* __restrict__ seed,
         const float* __restrict__ out_w, const float* __restrict__ out_b,
         const float* __restrict__ ff_w, const float* __restrict__ ff_b,
         const float* __restrict__ lnhg, const float* __restrict__ lnhb,
         const float* __restrict__ lnzg, const float* __restrict__ lnzb,
         float* __restrict__ out) {
  __shared__ float red[256];
  __shared__ float erow[256];
  const int s = blockIdx.x, t = threadIdx.x;

  float acc = out_b[t];
  const float* c = ctx + (size_t)s * DD;
  for (int k = 0; k < DD; ++k) acc += c[k] * out_w[(size_t)t * DD + k];
  const float hval = seed[(size_t)s * DD + t] + acc;

  red[t] = hval; __syncthreads();
  for (int off = 128; off; off >>= 1) { if (t < off) red[t] += red[t + off]; __syncthreads(); }
  const float mean = red[0] * (1.0f / DD); __syncthreads();
  const float dx = hval - mean;
  red[t] = dx * dx; __syncthreads();
  for (int off = 128; off; off >>= 1) { if (t < off) red[t] += red[t + off]; __syncthreads(); }
  const float var = red[0] * (1.0f / DD); __syncthreads();
  const float e = dx * rsqrtf(var + EPSV) * lnhg[t] + lnhb[t];

  erow[t] = e; __syncthreads();
  float f = ff_b[t];
  for (int k = 0; k < DD; ++k) f += erow[k] * ff_w[(size_t)t * DD + k];
  const float z = e + f;

  red[t] = z; __syncthreads();
  for (int off = 128; off; off >>= 1) { if (t < off) red[t] += red[t + off]; __syncthreads(); }
  const float mean2 = red[0] * (1.0f / DD); __syncthreads();
  const float dz = z - mean2;
  red[t] = dz * dz; __syncthreads();
  for (int off = 128; off; off >>= 1) { if (t < off) red[t] += red[t + off]; __syncthreads(); }
  const float var2 = red[0] * (1.0f / DD);
  out[(size_t)s * DD + t] = dz * rsqrtf(var2 + EPSV) * lnzg[t] + lnzb[t];
}

// ---------------------------------------------------------------- host launch
extern "C" void kernel_launch(void* const* d_in, const int* in_sizes, int n_in,
                              void* d_out, int out_size, void* d_ws, size_t ws_size,
                              hipStream_t stream) {
  (void)in_sizes; (void)n_in; (void)out_size; (void)ws_size;
  const float* x      = (const float*)d_in[0];
  const int*   ei     = (const int*)  d_in[1];
  const float* seed   = (const float*)d_in[2];
  const float* Wk     = (const float*)d_in[3];
  const float* bk     = (const float*)d_in[4];
  const float* Wv     = (const float*)d_in[5];
  const float* bv     = (const float*)d_in[6];
  const float* q_w    = (const float*)d_in[7];
  const float* out_w  = (const float*)d_in[8];
  const float* out_b  = (const float*)d_in[9];
  const float* ff_w   = (const float*)d_in[10];
  const float* ff_b   = (const float*)d_in[11];
  const float* ln_h_g = (const float*)d_in[12];
  const float* ln_h_b = (const float*)d_in[13];
  const float* ln_z_g = (const float*)d_in[14];
  const float* ln_z_b = (const float*)d_in[15];

  char* w = (char*)d_ws;
  size_t off = 0;
  auto alloc = [&](size_t bytes) { size_t o = off; off = (off + bytes + 255) & ~(size_t)255; return o; };
  unsigned short* xb   = (unsigned short*)(w + alloc((size_t)NN * DD * 2));
  unsigned short* wkT  = (unsigned short*)(w + alloc((size_t)DD * DD * 2));
  unsigned short* wvT  = (unsigned short*)(w + alloc((size_t)DD * DD * 2));
  float* hk    = (float*)(w + alloc((size_t)NN * DD * 4));
  float* hv    = (float*)(w + alloc((size_t)NN * DD * 4));
  float* kacc  = (float*)(w + alloc((size_t)NN * DD * 4));
  float* vacc  = (float*)(w + alloc((size_t)NN * DD * 4));
  unsigned short* kb = (unsigned short*)(w + alloc((size_t)NN * DD * 2));
  unsigned short* qb = (unsigned short*)(w + alloc((size_t)SS * DD * 2));
  float* deg   = (float*)(w + alloc((size_t)NN * 4));
  float* dinv  = (float*)(w + alloc((size_t)NN * 4));
  float* scores= (float*)(w + alloc((size_t)HH * SS * NN * 4));
  float* ctx   = (float*)(w + alloc((size_t)SS * DD * 4));

  // precision prep
  conv_x_bf16 <<<NN, 256, 0, stream>>>(x, xb);
  prep_weights<<<dim3(DD, 2), 256, 0, stream>>>(Wk, Wv, wkT, wvT);

  // degree normalization
  deg_init <<<(NN + 255) / 256, 256, 0, stream>>>(deg);
  deg_count<<<(EE + 255) / 256, 256, 0, stream>>>(ei, deg);
  deg_inv  <<<(NN + 255) / 256, 256, 0, stream>>>(deg, dinv);

  // dense GEMMs (WMMA)
  gemm_kv_wmma<<<NN / 16, 256, 0, stream>>>(xb, wkT, wvT, hk, hv);

  // GCN aggregation
  kv_init   <<<NN, 256, 0, stream>>>(hk, hv, bk, bv, dinv, kacc, vacc);
  spmm_edges<<<EE, 256, 0, stream>>>(ei, hk, hv, dinv, kacc, vacc);
  conv_k_bf16<<<NN, 256, 0, stream>>>(kacc, kb);

  // attention
  qproj      <<<SS, 256, 0, stream>>>(seed, q_w, qb);
  scores_wmma<<<dim3((NN / 16 + 7) / 8, HH), 256, 0, stream>>>(qb, kb, scores);
  softmax_ctx<<<HH * SS, 256, 0, stream>>>(scores, vacc, ctx);

  // epilogue
  epilogue<<<SS, 256, 0, stream>>>(ctx, seed, out_w, out_b, ff_w, ff_b,
                                   ln_h_g, ln_h_b, ln_z_g, ln_z_b, (float*)d_out);
}
